// LunarisMind_17428977287760
// MI455X (gfx1250) — matrix operand order repaired and checked by reading, not compiled, not run on hardware
//
#include <hip/hip_runtime.h>
#include <hip/hip_bf16.h>

typedef __attribute__((ext_vector_type(8)))  float  v8f;
typedef __attribute__((ext_vector_type(16))) __bf16 v16bf;

#define LAYERS 4
#define DMODEL 768
#define NHEAD  12
#define HDIM   64
#define SEQ    2048
#define RANK   32
#define DFFX2  6144
#define DFF    3072
#define VOCAB  32000
#define LORA_SCALE (1.0f / 32.0f)
#define NEG_INF (-1.0e9f)

// ---------------------------------------------------------------- helpers

__device__ __forceinline__ __bf16 f2bf(float f) {
    union { float f; unsigned u; } a; a.f = f;
    unsigned r = a.u + 0x7FFFu + ((a.u >> 16) & 1u);   // round-to-nearest-even
    union { unsigned short s; __bf16 b; } o; o.s = (unsigned short)(r >> 16);
    return o.b;
}

// A fragment: 16x32 bf16, element (m,k); p points at tile base, stride in
// elements. row must already include (lane & 15). Element s holds k = 16*(lane/16) + s (contiguous).
__device__ __forceinline__ v16bf frag_a(const __bf16* p, int stride, int row,
                                        int k0, int lane) {
    const __bf16* r = p + row * stride + k0;
    int kb = (lane & 16) ? 16 : 0;
    v16bf a;
#pragma unroll
    for (int i = 0; i < 8; ++i) {
        int k = kb + 2 * i;
        a[2 * i]     = r[k];
        a[2 * i + 1] = r[k + 1];
    }
    return a;
}

// B fragment: 32x16 bf16, element (k,n) = p[n*stride + k] (N-major storage).
// ncol must already include (lane & 15).
__device__ __forceinline__ v16bf frag_bn(const __bf16* p, int stride, int ncol,
                                         int k0, int lane) {
    const __bf16* r = p + ncol * stride + k0;
    int kb = (lane & 16) ? 16 : 0;
    v16bf b;
#pragma unroll
    for (int j = 0; j < 8; ++j) {
        int k = kb + 2 * j;
        b[2 * j]     = r[k];
        b[2 * j + 1] = r[k + 1];
    }
    return b;
}

// B fragment: element (k,n) = p[k*stride + n] (K-major storage).
__device__ __forceinline__ v16bf frag_bk(const __bf16* p, int stride, int k0,
                                         int ncol, int lane) {
    int kb = (lane & 16) ? 16 : 0;
    v16bf b;
#pragma unroll
    for (int j = 0; j < 8; ++j) {
        int k = k0 + kb + 2 * j;
        b[2 * j]     = p[k * stride + ncol];
        b[2 * j + 1] = p[(k + 1) * stride + ncol];
    }
    return b;
}

__device__ __forceinline__ float redmax16(float v) {
#pragma unroll
    for (int m = 8; m >= 1; m >>= 1) v = fmaxf(v, __shfl_xor(v, m, 32));
    return v;
}
__device__ __forceinline__ float redsum16(float v) {
#pragma unroll
    for (int m = 8; m >= 1; m >>= 1) v += __shfl_xor(v, m, 32);
    return v;
}

// ---------------------------------------------------------------- GEMM
// C[M,N] = alpha * A[M,K] x op(B) (+ Cadd)
// BT=true : B stored [N,K]  (nn.Linear weight, computes A @ B^T)
// BT=false: B stored [K,N]
// M % 128 == 0, K % 32 == 0, N ragged (masked).
// B (weights) is streamed with non-temporal loads: each weight is used once
// per forward, while A (activations, few MB) must stay resident in the 192MB
// L2 because every N-tile column of blocks re-reads it.
template <bool BT>
__global__ __launch_bounds__(256) void gemm_wmma_kernel(
    const float* __restrict__ A, const float* __restrict__ B,
    const float* __restrict__ Cadd, float* __restrict__ Cout,
    int M, int N, int K, float alpha, int nt_store) {
    __shared__ __bf16 As[128 * 34];
    __shared__ __bf16 Bs[128 * 34];

    const int tid = threadIdx.x, lane = tid & 31, wave = tid >> 5;
    const int m0 = blockIdx.y * 128, n0 = blockIdx.x * 128;
    const int wm = wave >> 2, wn = wave & 3;   // 2 x 4 waves -> 64 x 32 each
    const int lm = lane & 15;

    v8f acc[4][2] = {};

    for (int kt = 0; kt < K; kt += 32) {
        // stage A (f32 -> bf16)
#pragma unroll
        for (int i = 0; i < 16; ++i) {
            int idx = i * 256 + tid;
            int r = idx >> 5, c = idx & 31;
            As[r * 34 + c] = f2bf(A[(long)(m0 + r) * K + kt + c]);
        }
        // stage B (f32 -> bf16, non-temporal), N-major [n][k], zero-padded
        if (BT) {
#pragma unroll
            for (int i = 0; i < 16; ++i) {
                int idx = i * 256 + tid;
                int r = idx >> 5, c = idx & 31;
                int gn = n0 + r;
                float v = gn < N
                    ? __builtin_nontemporal_load(&B[(long)gn * K + kt + c])
                    : 0.0f;
                Bs[r * 34 + c] = f2bf(v);
            }
        } else {
#pragma unroll
            for (int i = 0; i < 16; ++i) {
                int idx = i * 256 + tid;
                int n = idx & 127, c = idx >> 7;
                int gn = n0 + n;
                float v = gn < N
                    ? __builtin_nontemporal_load(&B[(long)(kt + c) * N + gn])
                    : 0.0f;
                Bs[n * 34 + c] = f2bf(v);
            }
        }
        // prefetch the next K-slice while this slice's WMMAs run
        if (kt + 32 < K) {
            {   // A: 128 rows x 128B, two 64B halves per row
                int r = tid >> 1, half = (tid & 1) * 16;
                __builtin_prefetch(&A[(long)(m0 + r) * K + kt + 32 + half], 0, 1);
            }
            if (BT) {
                int r = tid >> 1, half = (tid & 1) * 16;
                int gn = n0 + r;
                if (gn < N)
                    __builtin_prefetch(&B[(long)gn * K + kt + 32 + half], 0, 1);
            } else {
                int r = tid >> 3, seg = (tid & 7) * 16;
                int gn = n0 + seg;
                if (gn < N)
                    __builtin_prefetch(&B[(long)(kt + 32 + r) * N + gn], 0, 1);
            }
        }
        __syncthreads();

        v16bf af[4];
#pragma unroll
        for (int tm = 0; tm < 4; ++tm)
            af[tm] = frag_a(As, 34, wm * 64 + tm * 16 + lm, 0, lane);
#pragma unroll
        for (int tn = 0; tn < 2; ++tn) {
            v16bf bf = frag_bn(Bs, 34, wn * 32 + tn * 16 + lm, 0, lane);
#pragma unroll
            for (int tm = 0; tm < 4; ++tm)
                acc[tm][tn] = __builtin_amdgcn_wmma_f32_16x16x32_bf16(
                    false, af[tm], false, bf, (short)0, acc[tm][tn], false,
                    false);
        }
        __syncthreads();
    }

    const int rb = (lane & 16) ? 8 : 0;
#pragma unroll
    for (int tm = 0; tm < 4; ++tm)
#pragma unroll
        for (int tn = 0; tn < 2; ++tn) {
            int col = n0 + wn * 32 + tn * 16 + lm;
            if (col < N) {
#pragma unroll
                for (int r = 0; r < 8; ++r) {
                    long row = m0 + wm * 64 + tm * 16 + rb + r;
                    float v = alpha * acc[tm][tn][r];
                    if (Cadd) v += Cadd[row * N + col];
                    if (nt_store)   // logits: written once, never re-read
                        __builtin_nontemporal_store(v, &Cout[row * N + col]);
                    else
                        Cout[row * N + col] = v;
                }
            }
        }
}

// ---------------------------------------------------------------- attention
// Flash-style causal attention with ALiBi. qkv: [S, 3*D] row-major.
// grid = (S/128, NHEAD), 256 threads (8 waves). Each wave owns 16 query rows,
// so softmax row-stats reduce within a 16-lane half-wave (wave32 native).
__global__ __launch_bounds__(256) void attn_kernel(
    const float* __restrict__ qkv, const float* __restrict__ slopes,
    float* __restrict__ ao) {
    __shared__ __bf16 Qs[128 * 65];       // [row][hd]
    __shared__ __bf16 Ks[64 * 65];        // [j][hd]
    __shared__ __bf16 Vs[64 * 65];        // [j][hd]
    __shared__ __bf16 Ps[8 * 16 * 65];    // per-wave 16 x 64 probs

    const int tid = threadIdx.x, lane = tid & 31, w = tid >> 5;
    const int it = blockIdx.x, head = blockIdx.y;
    const int i0 = it * 128;
    const int lm = lane & 15, rb = (lane & 16) ? 8 : 0;
    const float slope = slopes[head];
    const float scale = 0.125f;  // 1/sqrt(64)

    // stage Q tile (128 x 64)
#pragma unroll
    for (int i = 0; i < 32; ++i) {
        int idx = i * 256 + tid;
        int r = idx >> 6, c = idx & 63;
        Qs[r * 65 + c] = f2bf(qkv[(long)(i0 + r) * 2304 + head * 64 + c]);
    }
    __syncthreads();

    const int myrow0 = i0 + w * 16 + rb;  // + r gives the global query row
    float m_run[8], l_run[8];
    v8f o[4] = {};
#pragma unroll
    for (int r = 0; r < 8; ++r) { m_run[r] = -3.0e38f; l_run[r] = 0.0f; }

    v16bf aq0 = frag_a(Qs, 65, w * 16 + lm, 0, lane);
    v16bf aq1 = frag_a(Qs, 65, w * 16 + lm, 32, lane);

    const int njt = 2 * (it + 1);  // 64-wide key tiles covering j <= i0+127
    for (int jt = 0; jt < njt; ++jt) {
        const int j0 = jt * 64;
        // stage K, V tiles (64 x 64)
#pragma unroll
        for (int i = 0; i < 16; ++i) {
            int idx = i * 256 + tid;
            int r = idx >> 6, c = idx & 63;
            long base = (long)(j0 + r) * 2304 + head * 64 + c;
            Ks[r * 65 + c] = f2bf(qkv[base + 768]);
            Vs[r * 65 + c] = f2bf(qkv[base + 1536]);
        }
        __syncthreads();

        // scores: S = Q K^T  (16 x 64 per wave, 4 WMMA column tiles)
        v8f s[4];
#pragma unroll
        for (int nt = 0; nt < 4; ++nt) {
            v8f c = {};
            c = __builtin_amdgcn_wmma_f32_16x16x32_bf16(
                false, aq0, false, frag_bn(Ks, 65, nt * 16 + lm, 0, lane),
                (short)0, c, false, false);
            c = __builtin_amdgcn_wmma_f32_16x16x32_bf16(
                false, aq1, false, frag_bn(Ks, 65, nt * 16 + lm, 32, lane),
                (short)0, c, false, false);
            s[nt] = c;
        }

        // ALiBi + causal mask + online softmax (per query row r)
#pragma unroll
        for (int r = 0; r < 8; ++r) {
            const int gi = myrow0 + r;
            float mx = -3.0e38f;
#pragma unroll
            for (int nt = 0; nt < 4; ++nt) {
                int gj = j0 + nt * 16 + lm;
                int rel = gj - gi;
                float v = s[nt][r] * scale +
                          (rel <= 0 ? slope * (float)rel : NEG_INF);
                s[nt][r] = v;
                mx = fmaxf(mx, v);
            }
            mx = redmax16(mx);
            float mnew = fmaxf(m_run[r], mx);
            float corr = __expf(m_run[r] - mnew);
            float rs = 0.0f;
#pragma unroll
            for (int nt = 0; nt < 4; ++nt) {
                float p = __expf(s[nt][r] - mnew);
                s[nt][r] = p;
                rs += p;
            }
            rs = redsum16(rs);
            l_run[r] = l_run[r] * corr + rs;
            m_run[r] = mnew;
#pragma unroll
            for (int t = 0; t < 4; ++t) o[t][r] *= corr;
        }

        // C-layout -> A-layout via wave-private LDS
        __bf16* Pw = Ps + w * 16 * 65;
#pragma unroll
        for (int nt = 0; nt < 4; ++nt)
#pragma unroll
            for (int r = 0; r < 8; ++r)
                Pw[(rb + r) * 65 + nt * 16 + lm] = f2bf(s[nt][r]);

        // O += P V   (A = P 16x64, B = V 64x64 K-major)
#pragma unroll
        for (int kc = 0; kc < 2; ++kc) {
            v16bf ap = frag_a(Pw, 65, lm, kc * 32, lane);
#pragma unroll
            for (int t = 0; t < 4; ++t) {
                v16bf bv = frag_bk(Vs, 65, kc * 32, t * 16 + lm, lane);
                o[t] = __builtin_amdgcn_wmma_f32_16x16x32_bf16(
                    false, ap, false, bv, (short)0, o[t], false, false);
            }
        }
        __syncthreads();
    }

    // normalize and store
#pragma unroll
    for (int t = 0; t < 4; ++t) {
        int col = head * 64 + t * 16 + lm;
#pragma unroll
        for (int r = 0; r < 8; ++r) {
            long row = myrow0 + r;
            ao[row * DMODEL + col] = o[t][r] / l_run[r];
        }
    }
}

// ---------------------------------------------------------------- small ops

__global__ __launch_bounds__(256) void embed_kernel(
    const int* __restrict__ ids, const float* __restrict__ emb,
    float* __restrict__ x) {
    int s = blockIdx.x, tid = threadIdx.x;
    long row = (long)ids[s] * DMODEL;
    long dst = (long)s * DMODEL;
#pragma unroll
    for (int i = 0; i < 3; ++i) x[dst + tid + i * 256] = emb[row + tid + i * 256];
}

__global__ __launch_bounds__(256) void ln_kernel(
    const float* __restrict__ x, const float* __restrict__ g,
    const float* __restrict__ b, float* __restrict__ y) {
    __shared__ float red[256];
    int row = blockIdx.x, tid = threadIdx.x;
    const float* xr = x + (long)row * DMODEL;
    float v0 = xr[tid], v1 = xr[tid + 256], v2 = xr[tid + 512];
    red[tid] = v0 + v1 + v2;
    __syncthreads();
    for (int off = 128; off; off >>= 1) {
        if (tid < off) red[tid] += red[tid + off];
        __syncthreads();
    }
    float mean = red[0] * (1.0f / DMODEL);
    __syncthreads();
    float d0 = v0 - mean, d1 = v1 - mean, d2 = v2 - mean;
    red[tid] = d0 * d0 + d1 * d1 + d2 * d2;
    __syncthreads();
    for (int off = 128; off; off >>= 1) {
        if (tid < off) red[tid] += red[tid + off];
        __syncthreads();
    }
    float inv = rsqrtf(red[0] * (1.0f / DMODEL) + 1e-6f);
    float* yr = y + (long)row * DMODEL;
    yr[tid]       = d0 * inv * g[tid]       + b[tid];
    yr[tid + 256] = d1 * inv * g[tid + 256] + b[tid + 256];
    yr[tid + 512] = d2 * inv * g[tid + 512] + b[tid + 512];
}

// x += ls * y   grid (3, SEQ)
__global__ __launch_bounds__(256) void resid_kernel(
    float* __restrict__ x, const float* __restrict__ y,
    const float* __restrict__ ls) {
    int col = blockIdx.x * 256 + threadIdx.x;
    long i = (long)blockIdx.y * DMODEL + col;
    x[i] += ls[col] * y[i];
}

// ff = silu(h[:, :DFF]) * h[:, DFF:]   grid (12, SEQ)
__global__ __launch_bounds__(256) void swiglu_kernel(
    const float* __restrict__ h, float* __restrict__ ff) {
    int col = blockIdx.x * 256 + threadIdx.x;
    long row = blockIdx.y;
    float g = h[row * DFFX2 + col];
    float a = h[row * DFFX2 + DFF + col];
    ff[row * DFF + col] = (g / (1.0f + __expf(-g))) * a;
}

// ---------------------------------------------------------------- launch

extern "C" void kernel_launch(void* const* d_in, const int* in_sizes, int n_in,
                              void* d_out, int out_size, void* d_ws,
                              size_t ws_size, hipStream_t stream) {
    const int*   ids    = (const int*)d_in[0];
    const float* emb    = (const float*)d_in[1];
    const float* slopes = (const float*)d_in[2];
    const float* ln1_g  = (const float*)d_in[3];
    const float* ln1_b  = (const float*)d_in[4];
    const float* qkv_W  = (const float*)d_in[5];
    const float* qkv_A  = (const float*)d_in[6];
    const float* qkv_B  = (const float*)d_in[7];
    const float* out_W  = (const float*)d_in[8];
    const float* out_A  = (const float*)d_in[9];
    const float* out_B  = (const float*)d_in[10];
    const float* ln2_g  = (const float*)d_in[11];
    const float* ln2_b  = (const float*)d_in[12];
    const float* fc1_W  = (const float*)d_in[13];
    const float* fc1_A  = (const float*)d_in[14];
    const float* fc1_B  = (const float*)d_in[15];
    const float* fc2_W  = (const float*)d_in[16];
    const float* fc2_A  = (const float*)d_in[17];
    const float* fc2_B  = (const float*)d_in[18];
    const float* ls1    = (const float*)d_in[19];
    const float* ls2    = (const float*)d_in[20];
    const float* lnf_g  = (const float*)d_in[21];
    const float* lnf_b  = (const float*)d_in[22];

    float* ws    = (float*)d_ws;
    float* x     = ws;                              // [S, D]
    float* h     = x     + (size_t)SEQ * DMODEL;    // [S, D]
    float* t1    = h     + (size_t)SEQ * DMODEL;    // [S, R]
    float* delta = t1    + (size_t)SEQ * RANK;      // [S, 6144] (max N)
    float* buf   = delta + (size_t)SEQ * DFFX2;     // [S, 6144] qkv / hidden
    float* ao    = buf   + (size_t)SEQ * DFFX2;     // [S, D]
    float* ff    = ao    + (size_t)SEQ * DMODEL;    // [S, DFF]
    float* logits = (float*)d_out;                  // [S, V]

    const dim3 blk(256);
    auto gN = [](int N) { return (N + 127) / 128; };
    const int gM = SEQ / 128;

    embed_kernel<<<SEQ, blk, 0, stream>>>(ids, emb, x);

    for (int l = 0; l < LAYERS; ++l) {
        const float* qW = qkv_W + (size_t)l * 3 * DMODEL * DMODEL;
        const float* qA = qkv_A + (size_t)l * DMODEL * RANK;
        const float* qB = qkv_B + (size_t)l * RANK * 3 * DMODEL;
        const float* oW = out_W + (size_t)l * DMODEL * DMODEL;
        const float* oA = out_A + (size_t)l * DMODEL * RANK;
        const float* oB = out_B + (size_t)l * RANK * DMODEL;
        const float* f1W = fc1_W + (size_t)l * DFFX2 * DMODEL;
        const float* f1A = fc1_A + (size_t)l * DMODEL * RANK;
        const float* f1B = fc1_B + (size_t)l * RANK * DFFX2;
        const float* f2W = fc2_W + (size_t)l * DMODEL * DFF;
        const float* f2A = fc2_A + (size_t)l * DFF * RANK;
        const float* f2B = fc2_B + (size_t)l * RANK * DMODEL;

        // ---- attention block
        ln_kernel<<<SEQ, blk, 0, stream>>>(x, ln1_g + l * DMODEL,
                                           ln1_b + l * DMODEL, h);
        gemm_wmma_kernel<false><<<dim3(gN(RANK), gM), blk, 0, stream>>>(
            h, qA, nullptr, t1, SEQ, RANK, DMODEL, 1.0f, 0);
        gemm_wmma_kernel<false><<<dim3(gN(3 * DMODEL), gM), blk, 0, stream>>>(
            t1, qB, nullptr, delta, SEQ, 3 * DMODEL, RANK, LORA_SCALE, 0);
        gemm_wmma_kernel<true><<<dim3(gN(3 * DMODEL), gM), blk, 0, stream>>>(
            h, qW, delta, buf, SEQ, 3 * DMODEL, DMODEL, 1.0f, 0);

        attn_kernel<<<dim3(SEQ / 128, NHEAD), blk, 0, stream>>>(buf, slopes, ao);

        gemm_wmma_kernel<false><<<dim3(gN(RANK), gM), blk, 0, stream>>>(
            ao, oA, nullptr, t1, SEQ, RANK, DMODEL, 1.0f, 0);
        gemm_wmma_kernel<false><<<dim3(gN(DMODEL), gM), blk, 0, stream>>>(
            t1, oB, nullptr, delta, SEQ, DMODEL, RANK, LORA_SCALE, 0);
        gemm_wmma_kernel<true><<<dim3(gN(DMODEL), gM), blk, 0, stream>>>(
            ao, oW, delta, h, SEQ, DMODEL, DMODEL, 1.0f, 0);
        resid_kernel<<<dim3(3, SEQ), blk, 0, stream>>>(x, h, ls1 + l * DMODEL);

        // ---- SwiGLU FFN block
        ln_kernel<<<SEQ, blk, 0, stream>>>(x, ln2_g + l * DMODEL,
                                           ln2_b + l * DMODEL, h);
        gemm_wmma_kernel<false><<<dim3(gN(RANK), gM), blk, 0, stream>>>(
            h, f1A, nullptr, t1, SEQ, RANK, DMODEL, 1.0f, 0);
        gemm_wmma_kernel<false><<<dim3(gN(DFFX2), gM), blk, 0, stream>>>(
            t1, f1B, nullptr, delta, SEQ, DFFX2, RANK, LORA_SCALE, 0);
        gemm_wmma_kernel<true><<<dim3(gN(DFFX2), gM), blk, 0, stream>>>(
            h, f1W, delta, buf, SEQ, DFFX2, DMODEL, 1.0f, 0);

        swiglu_kernel<<<dim3(DFF / 256, SEQ), blk, 0, stream>>>(buf, ff);

        gemm_wmma_kernel<false><<<dim3(gN(RANK), gM), blk, 0, stream>>>(
            ff, f2A, nullptr, t1, SEQ, RANK, DFF, 1.0f, 0);
        gemm_wmma_kernel<false><<<dim3(gN(DMODEL), gM), blk, 0, stream>>>(
            t1, f2B, nullptr, delta, SEQ, DMODEL, RANK, LORA_SCALE, 0);
        gemm_wmma_kernel<true><<<dim3(gN(DMODEL), gM), blk, 0, stream>>>(
            ff, f2W, delta, h, SEQ, DMODEL, DFF, 1.0f, 0);
        resid_kernel<<<dim3(3, SEQ), blk, 0, stream>>>(x, h, ls2 + l * DMODEL);
    }

    // final LN + tied lm_head (logits streamed with NT stores: 262MB,
    // written once, never re-read on device)
    ln_kernel<<<SEQ, blk, 0, stream>>>(x, lnf_g, lnf_b, h);
    gemm_wmma_kernel<true><<<dim3(gN(VOCAB), gM), blk, 0, stream>>>(
        h, emb, nullptr, logits, SEQ, VOCAB, DMODEL, 1.0f, 1);
}
